// GRUGenerator_55585466745299
// MI455X (gfx1250) — matrix operand order repaired and checked by reading, not compiled, not run on hardware
//
#include <hip/hip_runtime.h>

// ---------------------------------------------------------------------------
// CDNA5 (gfx1250, wave32) mixed-precision GRU stack.
// - GEMMs on v_wmma_f32_16x16x32_bf16 (bf16 in, f32 accumulate)
// - global->LDS staging via GLOBAL_LOAD_ASYNC_TO_LDS_B128 (ASYNCcnt),
//   double-buffered so DMA of chunk k+1 overlaps WMMA of chunk k
// - all operand fragments loaded into distinct registers before the WMMA
//   batch: one dscnt wait per chunk, no WAR hazard NOPs between WMMAs
// - gate math + hidden-state master copy in f32
// ---------------------------------------------------------------------------

typedef __attribute__((ext_vector_type(16))) __bf16        v16bf;
typedef __attribute__((ext_vector_type(8)))  float         v8f;
typedef __attribute__((ext_vector_type(4)))  unsigned int  u32x4;

#define B_SZ     2048
#define H_SZ     512
#define D_INSZ   128
#define D_OUTSZ  64
#define T_STEPS  96

__device__ __forceinline__ unsigned short f2bf(float f) {
  unsigned u = __builtin_bit_cast(unsigned, f);
  u += 0x7FFFu + ((u >> 16) & 1u);            // round-to-nearest-even
  return (unsigned short)(u >> 16);
}

__device__ __forceinline__ v8f wmma_bf16(v16bf a, v16bf b, v8f c) {
  return __builtin_amdgcn_wmma_f32_16x16x32_bf16(false, a, false, b, (short)0, c,
                                                 false, false);
}

// Tiles in LDS are row-major, 32 bf16 (64 bytes) per row.
// A operand (16x32 MxK): lane<16 holds K{0..15}; lane>=16 K{16..31}.
__device__ __forceinline__ v16bf ld_fragA(const char* lds, int rowByte, int hsel) {
  union { u32x4 u[2]; v16bf v; } f;
  f.u[0] = *(const u32x4*)(lds + rowByte + hsel * 32);
  f.u[1] = *(const u32x4*)(lds + rowByte + hsel * 32 + 16);
  return f.v;
}
// B operand (32x16 KxN), row = output column: lane<16 holds K{0..15}; lane>=16 K{16..31}.
__device__ __forceinline__ v16bf ld_fragB(const char* lds, int rowByte, int hsel) {
  union { u32x4 u[2]; v16bf v; } f;
  f.u[0] = *(const u32x4*)(lds + rowByte + hsel * 32);
  f.u[1] = *(const u32x4*)(lds + rowByte + hsel * 32 + 16);
  return f.v;
}

// Async DMA: 16B per lane from (saddr + voff) to LDS byte address dst.
__device__ __forceinline__ void async_ld_b128(unsigned lds_dst, unsigned voff,
                                              const void* saddr) {
  asm volatile("global_load_async_to_lds_b128 %0, %1, %2"
               :: "v"(lds_dst), "v"(voff), "s"(saddr) : "memory");
}
__device__ __forceinline__ void wait_async0() {
  asm volatile("s_wait_asynccnt 0x0" ::: "memory");
}

__device__ __forceinline__ float sigm_f(float x) {
  float e = __expf(-x);
  return __builtin_amdgcn_rcpf(1.0f + e);
}
__device__ __forceinline__ float tanh_fast(float x) {
  x = fminf(fmaxf(x, -20.0f), 20.0f);
  float e = __expf(-2.0f * x);
  return (1.0f - e) * __builtin_amdgcn_rcpf(1.0f + e);
}

// ---------------------------------------------------------------------------
// Fused GRU cell. Block: 256 threads = 8 waves; output tile 64(M) x 64(N).
// Wave (mw, nw): 16 rows x 32 cols -> 2 col-subtiles x 3 gates x {i,h} = 12 accs.
// K chunks of 32 over [x-contribution | h-contribution], double-buffered LDS,
// async global->LDS staging (4 uniform b128 DMAs per thread per chunk).
// ---------------------------------------------------------------------------
__global__ __launch_bounds__(256) void gru_layer_kernel(
    const unsigned short* __restrict__ actx, int Kx,
    const unsigned short* __restrict__ acth,
    const float* __restrict__ hprevf,
    const unsigned short* __restrict__ Wih,   // (3H x Kx) bf16
    const unsigned short* __restrict__ Whh,   // (3H x H)  bf16
    const float* __restrict__ bih, const float* __restrict__ bhh,
    float* __restrict__ hnewf, unsigned short* __restrict__ hnewb) {
  __shared__ __align__(16) char sA[2][64 * 64];    // 64 rows x 32 bf16, x2 buf
  __shared__ __align__(16) char sW[2][192 * 64];   // 3 gates x 64 cols, x2 buf

  const int tid  = threadIdx.x;
  const int wave = tid >> 5, lane = tid & 31;
  const int mw = wave & 3, nw = wave >> 2;         // nw in {0,1}
  const int m0 = blockIdx.x * 64;
  const int n0 = blockIdx.y * 64;
  const int hsel = lane >> 4, lr = lane & 15;

  // ---- per-lane staging addresses (precomputed once) ----
  const int rA = tid >> 2, pAB = (tid & 3) * 16;
  const unsigned voffAx = (unsigned)(((m0 + rA) * Kx) * 2 + pAB);
  const unsigned voffAh = (unsigned)(((m0 + rA) * H_SZ) * 2 + pAB);
  const unsigned ldsA0  = (unsigned)(size_t)&sA[0][0];
  const unsigned dstA   = ldsA0 + (unsigned)(rA * 64 + pAB);

  unsigned voffWx[3], voffWh[3], dstW[3];
  const unsigned ldsW0 = (unsigned)(size_t)&sW[0][0];
#pragma unroll
  for (int j = 0; j < 3; ++j) {
    int idx = tid + j * 256;
    int r = idx >> 2, p = (idx & 3) * 16;
    int g = r >> 6, cc = r & 63;
    int wrow = g * H_SZ + n0 + cc;
    voffWx[j] = (unsigned)((wrow * Kx) * 2 + p);
    voffWh[j] = (unsigned)((wrow * H_SZ) * 2 + p);
    dstW[j]   = ldsW0 + (unsigned)(r * 64 + p);
  }

  const int nx = Kx >> 5;
  const int nc = nx + (H_SZ >> 5);

  auto prefetch = [&](int c, int par) {
    const unsigned kb = (unsigned)(((c < nx) ? c : (c - nx)) * 64);
    const unsigned pa = (unsigned)(par * (int)sizeof(sA[0]));
    const unsigned pw = (unsigned)(par * (int)sizeof(sW[0]));
    if (c < nx) {
      async_ld_b128(dstA + pa, voffAx + kb, actx);
#pragma unroll
      for (int j = 0; j < 3; ++j)
        async_ld_b128(dstW[j] + pw, voffWx[j] + kb, Wih);
    } else {
      async_ld_b128(dstA + pa, voffAh + kb, acth);
#pragma unroll
      for (int j = 0; j < 3; ++j)
        async_ld_b128(dstW[j] + pw, voffWh[j] + kb, Whh);
    }
  };

  v8f ai[3][2] = {}, ah[3][2] = {};

  prefetch(0, 0);
  for (int c = 0; c < nc; ++c) {
    const int par = c & 1;
    wait_async0();           // my chunk-c DMAs have landed in LDS
    __syncthreads();         // everyone's chunk-c data visible
    if (c + 1 < nc) prefetch(c + 1, par ^ 1);   // overlaps with WMMA below
    const char* bufA = sA[par];
    const char* bufW = sW[par];

    // Batch all operand fragments into distinct registers, then the 6 WMMAs.
    v16bf a = ld_fragA(bufA, (mw * 16 + lr) * 64, hsel);
    v16bf b[3][2];
#pragma unroll
    for (int g = 0; g < 3; ++g)
#pragma unroll
      for (int s = 0; s < 2; ++s)
        b[g][s] = ld_fragB(bufW, (g * 64 + nw * 32 + s * 16 + lr) * 64, hsel);

    if (c < nx) {
#pragma unroll
      for (int g = 0; g < 3; ++g)
#pragma unroll
        for (int s = 0; s < 2; ++s)
          ai[g][s] = wmma_bf16(a, b[g][s], ai[g][s]);
    } else {
#pragma unroll
      for (int g = 0; g < 3; ++g)
#pragma unroll
        for (int s = 0; s < 2; ++s)
          ah[g][s] = wmma_bf16(a, b[g][s], ah[g][s]);
    }
    __syncthreads();         // done reading buf[par]; safe to refill at c+2
  }

  // C/D layout: VGPR j -> row M = j + 8*hsel ; column N = lane&15.
  const int row0 = m0 + mw * 16 + hsel * 8;
#pragma unroll
  for (int s = 0; s < 2; ++s) {
    const int nG  = n0 + nw * 32 + s * 16 + lr;
    const float br  = bih[nG] + bhh[nG];
    const float bz  = bih[H_SZ + nG] + bhh[H_SZ + nG];
    const float bin = bih[2 * H_SZ + nG];
    const float bhn = bhh[2 * H_SZ + nG];
#pragma unroll
    for (int j = 0; j < 8; ++j) {
      const size_t idx = (size_t)(row0 + j) * H_SZ + nG;
      float r  = sigm_f(ai[0][s][j] + ah[0][s][j] + br);
      float z  = sigm_f(ai[1][s][j] + ah[1][s][j] + bz);
      float nn = tanh_fast(ai[2][s][j] + bin + r * (ah[2][s][j] + bhn));
      float hv = (1.0f - z) * nn + z * hprevf[idx];
      hnewf[idx] = hv;
      hnewb[idx] = f2bf(hv);
    }
  }
}

// ---------------------------------------------------------------------------
// Generic C = A @ W^T + bias, same async double-buffered pipeline.
// outb!=null: bf16 output (row stride ldo).  outb==null: f32 output written
// transposed as outf[b*Nt*Tt + n*Tt + tcol].
// ---------------------------------------------------------------------------
__global__ __launch_bounds__(256) void gemm_bias_kernel(
    const unsigned short* __restrict__ A, int K,
    const unsigned short* __restrict__ W,      // (N x K) bf16
    const float* __restrict__ bias,
    unsigned short* __restrict__ outb, int ldo,
    float* __restrict__ outf, int Nt, int tcol, int Tt) {
  __shared__ __align__(16) char sA[2][64 * 64];
  __shared__ __align__(16) char sW[2][64 * 64];

  const int tid  = threadIdx.x;
  const int wave = tid >> 5, lane = tid & 31;
  const int mw = wave & 3, nwv = wave >> 2;
  const int m0 = blockIdx.x * 64;
  const int n0 = blockIdx.y * 64;
  const int hsel = lane >> 4, lr = lane & 15;

  const int rA = tid >> 2, pAB = (tid & 3) * 16;
  const unsigned voffA = (unsigned)(((m0 + rA) * K) * 2 + pAB);
  const unsigned voffW = (unsigned)(((n0 + rA) * K) * 2 + pAB);
  const unsigned dstA  = (unsigned)(size_t)&sA[0][0] + (unsigned)(rA * 64 + pAB);
  const unsigned dstW  = (unsigned)(size_t)&sW[0][0] + (unsigned)(rA * 64 + pAB);

  const int nc = K >> 5;
  auto prefetch = [&](int c, int par) {
    const unsigned kb = (unsigned)(c * 64);
    const unsigned pb = (unsigned)(par * (int)sizeof(sA[0]));
    async_ld_b128(dstA + pb, voffA + kb, A);
    async_ld_b128(dstW + pb, voffW + kb, W);
  };

  v8f acc0 = {}, acc1 = {};
  prefetch(0, 0);
  for (int c = 0; c < nc; ++c) {
    const int par = c & 1;
    wait_async0();
    __syncthreads();
    if (c + 1 < nc) prefetch(c + 1, par ^ 1);
    const char* bufA = sA[par];
    const char* bufW = sW[par];
    v16bf a  = ld_fragA(bufA, (mw * 16 + lr) * 64, hsel);
    v16bf b0 = ld_fragB(bufW, (nwv * 32 + lr) * 64, hsel);
    v16bf b1 = ld_fragB(bufW, (nwv * 32 + 16 + lr) * 64, hsel);
    acc0 = wmma_bf16(a, b0, acc0);
    acc1 = wmma_bf16(a, b1, acc1);
    __syncthreads();
  }

  const int nA = n0 + nwv * 32 + lr;
  const int nB = nA + 16;
  const float bA = bias[nA], bB = bias[nB];
  const int row0 = m0 + mw * 16 + hsel * 8;
  if (outb) {
#pragma unroll
    for (int j = 0; j < 8; ++j) {
      int b = row0 + j;
      outb[(size_t)b * ldo + nA] = f2bf(acc0[j] + bA);
      outb[(size_t)b * ldo + nB] = f2bf(acc1[j] + bB);
    }
  } else {
#pragma unroll
    for (int j = 0; j < 8; ++j) {
      int b = row0 + j;
      outf[((size_t)b * Nt + nA) * Tt + tcol] = acc0[j] + bA;
      outf[((size_t)b * Nt + nB) * Tt + tcol] = acc1[j] + bB;
    }
  }
}

__global__ void cvt_f32_bf16_kernel(const float* __restrict__ s,
                                    unsigned short* __restrict__ d, int n) {
  int i = blockIdx.x * blockDim.x + threadIdx.x;
  if (i < n) d[i] = f2bf(s[i]);
}

__global__ void fill_zero_kernel(unsigned int* __restrict__ p, int n_u32) {
  int i = blockIdx.x * blockDim.x + threadIdx.x;
  if (i < n_u32) p[i] = 0u;
}

// ---------------------------------------------------------------------------
extern "C" void kernel_launch(void* const* d_in, const int* in_sizes, int n_in,
                              void* d_out, int out_size, void* d_ws, size_t ws_size,
                              hipStream_t stream) {
  (void)in_sizes; (void)n_in; (void)out_size; (void)ws_size;

  const float* x      = (const float*)d_in[0];
  const float* W_ih0  = (const float*)d_in[1];
  const float* W_hh0  = (const float*)d_in[2];
  const float* b_ih0  = (const float*)d_in[3];
  const float* b_hh0  = (const float*)d_in[4];
  const float* W_ih1  = (const float*)d_in[5];
  const float* W_hh1  = (const float*)d_in[6];
  const float* b_ih1  = (const float*)d_in[7];
  const float* b_hh1  = (const float*)d_in[8];
  const float* W_fc   = (const float*)d_in[9];
  const float* b_fc   = (const float*)d_in[10];
  const float* W_proj = (const float*)d_in[11];
  const float* b_proj = (const float*)d_in[12];

  char* w = (char*)d_ws;
  auto take = [&](size_t bytes) {
    char* p = w;
    w += (bytes + 255) & ~(size_t)255;
    return p;
  };

  unsigned short* wih0  = (unsigned short*)take((size_t)3 * H_SZ * D_INSZ * 2);
  unsigned short* whh0  = (unsigned short*)take((size_t)3 * H_SZ * H_SZ * 2);
  unsigned short* wih1  = (unsigned short*)take((size_t)3 * H_SZ * H_SZ * 2);
  unsigned short* whh1  = (unsigned short*)take((size_t)3 * H_SZ * H_SZ * 2);
  unsigned short* wfc   = (unsigned short*)take((size_t)D_INSZ * H_SZ * 2);
  unsigned short* wproj = (unsigned short*)take((size_t)D_OUTSZ * D_INSZ * 2);
  float* h0f[2] = {(float*)take((size_t)B_SZ * H_SZ * 4),
                   (float*)take((size_t)B_SZ * H_SZ * 4)};
  float* h1f[2] = {(float*)take((size_t)B_SZ * H_SZ * 4),
                   (float*)take((size_t)B_SZ * H_SZ * 4)};
  unsigned short* h0b[2] = {(unsigned short*)take((size_t)B_SZ * H_SZ * 2),
                            (unsigned short*)take((size_t)B_SZ * H_SZ * 2)};
  unsigned short* h1b[2] = {(unsigned short*)take((size_t)B_SZ * H_SZ * 2),
                            (unsigned short*)take((size_t)B_SZ * H_SZ * 2)};
  unsigned short* xbuf = (unsigned short*)take((size_t)B_SZ * D_INSZ * 2);

  auto cvt = [&](const float* s, unsigned short* d, int n) {
    cvt_f32_bf16_kernel<<<(n + 255) / 256, 256, 0, stream>>>(s, d, n);
  };
  cvt(W_ih0, wih0, 3 * H_SZ * D_INSZ);
  cvt(W_hh0, whh0, 3 * H_SZ * H_SZ);
  cvt(W_ih1, wih1, 3 * H_SZ * H_SZ);
  cvt(W_hh1, whh1, 3 * H_SZ * H_SZ);
  cvt(W_fc, wfc, D_INSZ * H_SZ);
  cvt(W_proj, wproj, D_OUTSZ * D_INSZ);
  cvt(x, xbuf, B_SZ * D_INSZ);   // x is (B,1,128) contiguous

  auto zero = [&](void* p, size_t bytes) {
    int n = (int)(bytes / 4);
    fill_zero_kernel<<<(n + 255) / 256, 256, 0, stream>>>((unsigned int*)p, n);
  };
  zero(h0f[0], (size_t)B_SZ * H_SZ * 4);
  zero(h1f[0], (size_t)B_SZ * H_SZ * 4);
  zero(h0b[0], (size_t)B_SZ * H_SZ * 2);
  zero(h1b[0], (size_t)B_SZ * H_SZ * 2);

  const dim3 blk(256);
  const dim3 gGru(B_SZ / 64, H_SZ / 64);
  const dim3 gFc(B_SZ / 64, D_INSZ / 64);
  const dim3 gPr(B_SZ / 64, D_OUTSZ / 64);

  int cur = 0;
  for (int t = 0; t < T_STEPS; ++t) {
    gru_layer_kernel<<<gGru, blk, 0, stream>>>(
        xbuf, D_INSZ, h0b[cur], h0f[cur], wih0, whh0, b_ih0, b_hh0,
        h0f[cur ^ 1], h0b[cur ^ 1]);
    gru_layer_kernel<<<gGru, blk, 0, stream>>>(
        h0b[cur ^ 1], H_SZ, h1b[cur], h1f[cur], wih1, whh1, b_ih1, b_hh1,
        h1f[cur ^ 1], h1b[cur ^ 1]);
    gemm_bias_kernel<<<gFc, blk, 0, stream>>>(
        h1b[cur ^ 1], H_SZ, wfc, b_fc, xbuf, D_INSZ, nullptr, 0, 0, 0);
    gemm_bias_kernel<<<gPr, blk, 0, stream>>>(
        xbuf, D_INSZ, wproj, b_proj, nullptr, 0, (float*)d_out, D_OUTSZ, t,
        T_STEPS);
    cur ^= 1;
  }
}
